// EyeRobotAgent_69157563400910
// MI455X (gfx1250) — matrix operand order repaired and checked by reading, not compiled, not run on hardware
//
#include <hip/hip_runtime.h>

// CDNA5 / gfx1250 flash-attention with eye-block mask.
// S^T = K x Q^T (wmma f32_16x16x32_f16), column-softmax (per-lane stats),
// O^T += V^T x P^T. f16 operands, f32 accumulate.
// Fast path (if ws fits): pre-pass kernels build f16 K, f16 transposed V and a
// per-(kvblock,q) mask-bit table in d_ws; main loop is then b128 loads + WMMA.

typedef _Float16 v16h __attribute__((ext_vector_type(16)));
typedef __fp16   h2   __attribute__((ext_vector_type(2)));   // cvt_pkrtz result type
typedef float    v8f  __attribute__((ext_vector_type(8)));

#define S_LEN       3456
#define DHEAD       32
#define NBH         24      // B*H
#define NKVB        108     // S_LEN/32
#define FPT         216
#define IMG_START   20      // FPT - img_feat_size(196)
#define JOINT_START 4       // IMG_START - act_size(16)
#define PROPRIO     4
#define NEG_INF     (-__builtin_inff())

__device__ __forceinline__ unsigned div216(unsigned x) {
  // exact floor(x/216) for x < ~20000
  return (x * 19419u) >> 22;
}

// Eye-block mask predicate (matches reference eye_mask).
__device__ __forceinline__ bool eye_allow(unsigned q_t, bool q_img, bool q_joint,
                                          unsigned kv) {
  const unsigned t  = div216(kv);
  const unsigned mm = kv - t * FPT;
  const int dt      = (int)q_t - (int)t;
  const bool causal = (unsigned)dt < 8u;
  const bool kv_img = mm >= IMG_START;
  const bool kv_pro = mm == PROPRIO;
  const bool kv_jnt = (mm - JOINT_START) < 16u;
  const bool blocked = (dt == 0) ? (q_img && kv_img)
                                 : (kv_img || kv_pro || (q_joint && kv_jnt));
  return causal && !blocked;
}

union HPack { h2 h[8]; v16h v; };

// ---- slow-path loaders (f32 global -> f16 regs) ---------------------------
__device__ __forceinline__ v16h load_A_row(const float* __restrict__ rowptr, int dbase) {
  const float4* p0 = (const float4*)(rowptr + dbase);
  const float4* p1 = (const float4*)(rowptr + dbase + 16);
  float4 a0 = p0[0], a1 = p0[1], a2 = p1[0], a3 = p1[1];
  HPack u;
  u.h[0] = __builtin_amdgcn_cvt_pkrtz(a0.x, a0.y);
  u.h[1] = __builtin_amdgcn_cvt_pkrtz(a0.z, a0.w);
  u.h[2] = __builtin_amdgcn_cvt_pkrtz(a1.x, a1.y);
  u.h[3] = __builtin_amdgcn_cvt_pkrtz(a1.z, a1.w);
  u.h[4] = __builtin_amdgcn_cvt_pkrtz(a2.x, a2.y);
  u.h[5] = __builtin_amdgcn_cvt_pkrtz(a2.z, a2.w);
  u.h[6] = __builtin_amdgcn_cvt_pkrtz(a3.x, a3.y);
  u.h[7] = __builtin_amdgcn_cvt_pkrtz(a3.z, a3.w);
  return u.v;
}

__device__ __forceinline__ v16h load_A_vt_gather(const float* __restrict__ vbase,
                                                 int d, int hi) {
  const float* p  = vbase + (size_t)(hi * 8) * DHEAD + d;
  const float* p2 = p + 16 * DHEAD;
  HPack u;
#pragma unroll
  for (int j = 0; j < 4; ++j)
    u.h[j] = __builtin_amdgcn_cvt_pkrtz(p[(2 * j) * DHEAD], p[(2 * j + 1) * DHEAD]);
#pragma unroll
  for (int j = 0; j < 4; ++j)
    u.h[4 + j] = __builtin_amdgcn_cvt_pkrtz(p2[(2 * j) * DHEAD], p2[(2 * j + 1) * DHEAD]);
  return u.v;
}

// ---- fast-path loader: two 16B chunks of pre-packed f16 -------------------
__device__ __forceinline__ v16h load2x8h(const _Float16* __restrict__ p) {
  union { uint4 u[2]; v16h v; } u;
  u.u[0] = *(const uint4*)p;
  u.u[1] = *(const uint4*)(p + 16);
  return u.v;
}

// ===========================================================================
// Pre-pass kernels (fast path only)
// ===========================================================================

// One u32 of mask bits per (kvblock, q): bit b = allow(q, kvblock*32+b).
__global__ __launch_bounds__(256) void eye_mask_words(unsigned* __restrict__ mw) {
  const int w = blockIdx.x * 256 + threadIdx.x;
  if (w >= NKVB * S_LEN) return;
  const int kvb = w / S_LEN;
  const int q   = w - kvb * S_LEN;
  const unsigned q_t = div216((unsigned)q);
  const unsigned q_m = (unsigned)q - q_t * FPT;
  const bool q_img   = q_m >= IMG_START;
  const bool q_joint = (q_m - JOINT_START) < 16u;
  unsigned word = 0;
#pragma unroll
  for (int b = 0; b < 32; ++b)
    word |= (unsigned)eye_allow(q_t, q_img, q_joint, (unsigned)(kvb * 32 + b)) << b;
  mw[w] = word;
}

// K (f32) -> Kh (f16), same [bh][kv][d] layout. 4 floats / thread.
__global__ __launch_bounds__(256) void cvt_k_f16(const float* __restrict__ K,
                                                 _Float16* __restrict__ Kh) {
  const size_t i = ((size_t)blockIdx.x * 256 + threadIdx.x) * 4;
  if (i >= (size_t)NBH * S_LEN * DHEAD) return;
  float4 f = *(const float4*)(K + i);
  union { h2 h[2]; uint2 u; } u;
  u.h[0] = __builtin_amdgcn_cvt_pkrtz(f.x, f.y);
  u.h[1] = __builtin_amdgcn_cvt_pkrtz(f.z, f.w);
  *(uint2*)(Kh + i) = u.u;
}

// V (f32 [bh][kv][d]) -> Vth (f16 [bh][d][kv]). One thread per kv row:
// reads its 128B row once, writes 32 lane-coalesced b16 columns.
__global__ __launch_bounds__(256) void transpose_v_f16(const float* __restrict__ V,
                                                       _Float16* __restrict__ Vth) {
  const int i = blockIdx.x * 256 + threadIdx.x;   // 0 .. NBH*S_LEN-1
  if (i >= NBH * S_LEN) return;
  const int kv = i % S_LEN;
  const int bh = i / S_LEN;
  const float* src = V + (size_t)i * DHEAD;
  _Float16* dst = Vth + (size_t)bh * DHEAD * S_LEN + kv;
#pragma unroll
  for (int d = 0; d < DHEAD; ++d)
    dst[(size_t)d * S_LEN] = (_Float16)src[d];
}

// ===========================================================================
// Main attention kernel (templated on path)
// ===========================================================================
template <bool FAST>
__global__ __launch_bounds__(256) void eye_attn_kernel(
    const float* __restrict__ Q, const float* __restrict__ K,
    const float* __restrict__ V, float* __restrict__ O,
    const _Float16* __restrict__ Kh, const _Float16* __restrict__ Vth,
    const unsigned* __restrict__ MW) {
  const int lane = threadIdx.x & 31;
  const int wave = threadIdx.x >> 5;
  const int l16  = lane & 15;
  const int hi   = lane >> 4;              // 0 or 1

  const int bh    = blockIdx.y;
  const int qtile = blockIdx.x * 8 + wave; // 27 blocks x 8 waves = 216 tiles
  const int q0    = qtile * 16;

  const size_t base = (size_t)bh * S_LEN * DHEAD;
  const float* Qb = Q + base;
  const float* Kb = K + base;
  const float* Vb = V + base;
  float*       Ob = O + base;
  const _Float16* Khb  = Kh + base;
  const _Float16* Vthb = Vth + base;       // [d][kv] within this bh

  // Per-lane query flags (lane's q column is fixed for the whole kernel)
  const int q        = q0 + l16;
  const unsigned q_t = div216((unsigned)q);
  const unsigned q_m = (unsigned)q - q_t * FPT;
  const bool q_img   = q_m >= IMG_START;
  const bool q_joint = (q_m - JOINT_START) < 16u;

  // Q^T B-operand: lane = column q, half h holds d 8h..8h+7 and 16+8h..16+8h+7.
  // Fold softmax scale * log2(e) into Q so softmax uses exp2.
  const float kScale = 0.17677669529663689f * 1.4426950408889634f;
  const float* qptr = Qb + (size_t)q * DHEAD + hi * 8;
  float4 f0 = ((const float4*)qptr)[0];
  float4 f1 = ((const float4*)qptr)[1];
  float4 f2 = ((const float4*)qptr)[4];
  float4 f3 = ((const float4*)qptr)[5];
  HPack uq;
  uq.h[0] = __builtin_amdgcn_cvt_pkrtz(f0.x * kScale, f0.y * kScale);
  uq.h[1] = __builtin_amdgcn_cvt_pkrtz(f0.z * kScale, f0.w * kScale);
  uq.h[2] = __builtin_amdgcn_cvt_pkrtz(f1.x * kScale, f1.y * kScale);
  uq.h[3] = __builtin_amdgcn_cvt_pkrtz(f1.z * kScale, f1.w * kScale);
  uq.h[4] = __builtin_amdgcn_cvt_pkrtz(f2.x * kScale, f2.y * kScale);
  uq.h[5] = __builtin_amdgcn_cvt_pkrtz(f2.z * kScale, f2.w * kScale);
  uq.h[6] = __builtin_amdgcn_cvt_pkrtz(f3.x * kScale, f3.y * kScale);
  uq.h[7] = __builtin_amdgcn_cvt_pkrtz(f3.z * kScale, f3.w * kScale);
  const v16h bQ = uq.v;

  // KV range from the causal window: kv_t in [q_t-7, q_t]
  const unsigned qt_min = div216((unsigned)q0);
  const unsigned qt_max = div216((unsigned)(q0 + 15));
  int kv_start = ((int)qt_min - 7);
  if (kv_start < 0) kv_start = 0;
  kv_start = (kv_start * FPT) & ~31;
  int kv_end = (int)(qt_max + 1) * FPT;
  if (kv_end > S_LEN) kv_end = S_LEN;
  kv_end = (kv_end + 31) & ~31;            // S_LEN is a multiple of 32

  // Flash state: per-lane scalars (row = lane's q)
  float m_i = -1.0e30f;
  float l_i = 0.0f;
  v8f o0 = {};
  v8f o1 = {};
  const v8f zc = {};

  for (int kv0 = kv_start; kv0 < kv_end; kv0 += 32) {
    // --- S^T = K x Q^T ------------------------------------------------
    v16h aK0, aK1;
    if constexpr (FAST) {
      const _Float16* kr0 = Khb + (size_t)(kv0 + l16) * DHEAD + hi * 8;
      aK0 = load2x8h(kr0);
      aK1 = load2x8h(kr0 + 16 * DHEAD);
    } else {
      aK0 = load_A_row(Kb + (size_t)(kv0 + l16) * DHEAD, hi * 8);
      aK1 = load_A_row(Kb + (size_t)(kv0 + 16 + l16) * DHEAD, hi * 8);
    }
    v8f st0 = __builtin_amdgcn_wmma_f32_16x16x32_f16(false, aK0, false, bQ,
                                                     (short)0, zc, false, false);
    v8f st1 = __builtin_amdgcn_wmma_f32_16x16x32_f16(false, aK1, false, bQ,
                                                     (short)0, zc, false, false);

    // --- mask + row max ----------------------------------------------
    float s0[8], s1[8];
    float mx = NEG_INF;
    if constexpr (FAST) {
      const unsigned word =
          MW[(size_t)(kv0 >> 5) * S_LEN + q] >> (8 * hi);
#pragma unroll
      for (int p = 0; p < 8; ++p) {
        s0[p] = ((word >> p) & 1u) ? st0[p] : NEG_INF;
        s1[p] = ((word >> (16 + p)) & 1u) ? st1[p] : NEG_INF;
        mx = fmaxf(mx, fmaxf(s0[p], s1[p]));
      }
    } else {
#pragma unroll
      for (int p = 0; p < 8; ++p) {
        const unsigned kva = (unsigned)(kv0 + p + 8 * hi);
        const unsigned kvb = kva + 16u;
        s0[p] = eye_allow(q_t, q_img, q_joint, kva) ? st0[p] : NEG_INF;
        s1[p] = eye_allow(q_t, q_img, q_joint, kvb) ? st1[p] : NEG_INF;
        mx = fmaxf(mx, fmaxf(s0[p], s1[p]));
      }
    }
    mx = fmaxf(mx, __shfl_xor(mx, 16));      // combine kv lane-halves

    // --- online softmax (log2 domain) --------------------------------
    const float m_new = fmaxf(m_i, mx);
    const float alpha = __builtin_amdgcn_exp2f(m_i - m_new);
    float p0[8], p1[8];
    float rs = 0.0f;
#pragma unroll
    for (int p = 0; p < 8; ++p) {
      p0[p] = __builtin_amdgcn_exp2f(s0[p] - m_new);
      p1[p] = __builtin_amdgcn_exp2f(s1[p] - m_new);
      rs += p0[p] + p1[p];
    }
    rs += __shfl_xor(rs, 16);
    l_i = l_i * alpha + rs;
    m_i = m_new;
#pragma unroll
    for (int i = 0; i < 8; ++i) { o0[i] *= alpha; o1[i] *= alpha; }

    // --- P^T B-operand: pack own p0 (kv 8h..) then own p1 (kv 16+8h..) ----
    HPack up;
#pragma unroll
    for (int r = 0; r < 4; ++r) {
      up.h[r] = __builtin_amdgcn_cvt_pkrtz(p0[2 * r], p0[2 * r + 1]);
    }
#pragma unroll
    for (int r = 4; r < 8; ++r) {
      const int s = 2 * r - 8;
      up.h[r] = __builtin_amdgcn_cvt_pkrtz(p1[s], p1[s + 1]);
    }
    const v16h bP = up.v;

    // --- O^T += V^T x P^T --------------------------------------------
    v16h aV0, aV1;
    if constexpr (FAST) {
      const _Float16* vr0 = Vthb + (size_t)l16 * S_LEN + kv0 + hi * 8;
      aV0 = load2x8h(vr0);
      aV1 = load2x8h(vr0 + (size_t)16 * S_LEN);
    } else {
      const float* vblk = Vb + (size_t)kv0 * DHEAD;
      aV0 = load_A_vt_gather(vblk, l16, hi);
      aV1 = load_A_vt_gather(vblk, 16 + l16, hi);
    }
    o0 = __builtin_amdgcn_wmma_f32_16x16x32_f16(false, aV0, false, bP,
                                                (short)0, o0, false, false);
    o1 = __builtin_amdgcn_wmma_f32_16x16x32_f16(false, aV1, false, bP,
                                                (short)0, o1, false, false);
  }

  // --- normalize and store (O^T layout: lane = q, vgpr/half = d) ------
  const float inv = 1.0f / l_i;
  float* optr = Ob + (size_t)q * DHEAD;
#pragma unroll
  for (int p = 0; p < 8; ++p) {
    optr[p + 8 * hi]      = o0[p] * inv;
    optr[16 + p + 8 * hi] = o1[p] * inv;
  }
}

// ===========================================================================
extern "C" void kernel_launch(void* const* d_in, const int* in_sizes, int n_in,
                              void* d_out, int out_size, void* d_ws, size_t ws_size,
                              hipStream_t stream) {
  const float* Q = (const float*)d_in[0];
  const float* K = (const float*)d_in[1];
  const float* V = (const float*)d_in[2];
  float* O = (float*)d_out;
  (void)in_sizes; (void)n_in; (void)out_size;

  constexpr size_t MW_BYTES = (size_t)NKVB * S_LEN * 4;          // 1.49 MB
  constexpr size_t KH_OFF   = (MW_BYTES + 255) & ~(size_t)255;
  constexpr size_t KH_BYTES = (size_t)NBH * S_LEN * DHEAD * 2;   // 5.3 MB
  constexpr size_t VT_OFF   = KH_OFF + KH_BYTES;
  constexpr size_t NEED     = VT_OFF + KH_BYTES;                 // ~12.1 MB

  const dim3 grid(S_LEN / 16 / 8, NBH);    // 27 x 24, 8 waves/block

  if (ws_size >= NEED) {
    unsigned*  MW  = (unsigned*)d_ws;
    _Float16*  Kh  = (_Float16*)((char*)d_ws + KH_OFF);
    _Float16*  Vth = (_Float16*)((char*)d_ws + VT_OFF);
    eye_mask_words<<<(NKVB * S_LEN + 255) / 256, 256, 0, stream>>>(MW);
    cvt_k_f16<<<(int)(((size_t)NBH * S_LEN * DHEAD / 4 + 255) / 256), 256, 0, stream>>>(K, Kh);
    transpose_v_f16<<<(NBH * S_LEN + 255) / 256, 256, 0, stream>>>(V, Vth);
    eye_attn_kernel<true><<<grid, 256, 0, stream>>>(Q, K, V, O, Kh, Vth, MW);
  } else {
    eye_attn_kernel<false><<<grid, 256, 0, stream>>>(Q, K, V, O,
                                                     (const _Float16*)nullptr,
                                                     (const _Float16*)nullptr,
                                                     (const unsigned*)nullptr);
  }
}
